// DPVFAConvBlock_81647328297405
// MI455X (gfx1250) — matrix operand order repaired and checked by reading, not compiled, not run on hardware
//
#include <hip/hip_runtime.h>
#include <hip/hip_bf16.h>

// ---------------------------------------------------------------------------
// Geometry / constants
// ---------------------------------------------------------------------------
#define V    64
#define V3   262144          // 64^3  == 1<<18
#define PV   66
#define PV3  287496          // 66^3
#define CCH  32
#define CEMB 16
#define GN_N 2097152.0f      // 8 ch * 64^3 elements per group
#define NEG_SLOPE 0.2f
#define GEPS 1e-5f

// conv kernel LDS plan (dynamic shared memory):
//   wlds  : 27648 bf16 = 55296 B   (swizzled A-fragments, 27 taps x 2 mtiles)
//   patch : 38016 bf16 = 76032 B   (3 z-planes x 6 y-rows x 66 x x 32 ch)
#define WLDS_HALFS   27648
#define PATCH_HALFS  38016
#define CONV_SMEM_BYTES ((WLDS_HALFS + PATCH_HALFS) * 2)

typedef __attribute__((ext_vector_type(16))) __bf16 v16bf;
typedef __attribute__((ext_vector_type(8)))  float  v8f;

// ---------------------------------------------------------------------------
// CDNA5 async global->LDS helpers (inline asm: portable across toolchains)
// ---------------------------------------------------------------------------
__device__ __forceinline__ unsigned lds_byte_off(const void* p) {
    // generic -> addrspace(3) -> 32-bit LDS byte offset
    return (unsigned)(unsigned long long)(__attribute__((address_space(3))) const char*)p;
}

__device__ __forceinline__ void async_load_b128(unsigned lds_off, const void* gaddr) {
    asm volatile("global_load_async_to_lds_b128 %0, %1, off"
                 :: "v"(lds_off), "v"((unsigned long long)gaddr)
                 : "memory");
}

__device__ __forceinline__ void wait_asynccnt0() {
    asm volatile("s_wait_asynccnt 0x0" ::: "memory");
}

// ---------------------------------------------------------------------------
// 1) Cast fp32 NCDHW input -> bf16 zero-padded channel-last [img][66^3][32]
// ---------------------------------------------------------------------------
__global__ void cast_pad_kernel(const float* __restrict__ moving,
                                const float* __restrict__ fixedf,
                                __bf16* __restrict__ xpad) {
    long long idx = (long long)blockIdx.x * blockDim.x + threadIdx.x;
    if (idx >= 2LL * PV3 * CCH) return;
    int ci = (int)(idx & 31);
    long long pv = idx >> 5;
    int img = (int)(pv / PV3);
    int p = (int)(pv % PV3);
    int x = p % PV; int t = p / PV; int y = t % PV; int z = t / PV;
    float val = 0.0f;
    if (x >= 1 && x <= V && y >= 1 && y <= V && z >= 1 && z <= V) {
        const float* src = (img == 0) ? moving : fixedf;
        val = src[((long long)ci << 18) + (((z - 1) * V + (y - 1)) * V + (x - 1))];
    }
    xpad[idx] = (__bf16)val;
}

// ---------------------------------------------------------------------------
// 2) Swizzle conv weights [32][32][27] fp32 -> WMMA A-fragment layout bf16
//    wfrag[t][mtile][lane][e], contiguous k order, same as the B operand:
//    m = lane%16 ; k = 16*(lane>>4) + e
// ---------------------------------------------------------------------------
__global__ void prep_weights_kernel(const float* __restrict__ w,
                                    __bf16* __restrict__ wfrag) {
    int idx = blockIdx.x * blockDim.x + threadIdx.x;   // 27*2*32*16 = 27648
    if (idx >= 27648) return;
    int e    = idx & 15;
    int lane = (idx >> 4) & 31;
    int mt   = (idx >> 9) & 1;
    int t    = idx >> 10;
    int hi   = lane >> 4;
    int m    = lane & 15;
    int j    = e >> 1;
    int k    = 16 * hi + e; (void)j;
    int co   = mt * 16 + m;
    wfrag[idx] = (__bf16)w[(co * CCH + k) * 27 + t];
}

// ---------------------------------------------------------------------------
// 3) Implicit-GEMM 3x3x3 conv, 32->32, via v_wmma_f32_16x16x32_bf16.
//    Block = 8 waves / 256 threads; owns a (1 z) x (4 y) x (64 x) output slab.
//    Weights (55 KB) + input patch (76 KB, 3z x 6y x 66x x 32ch) staged into
//    LDS with global_load_async_to_lds_b128; each wave computes 2 tiles of
//    16 voxels x 32 co -> 27 taps x 4 WMMAs = 108 WMMAs per wave, all
//    operands served from LDS.
// ---------------------------------------------------------------------------
__global__ void __launch_bounds__(256)
conv3_wmma_kernel(const __bf16* __restrict__ xpad,
                  const __bf16* __restrict__ wfrag,
                  const float* __restrict__ bias,
                  float* __restrict__ hout) {
    extern __shared__ __align__(32) __bf16 smem[];
    __bf16* wlds  = smem;               // 27648 halfs
    __bf16* patch = smem + WLDS_HALFS;  // 38016 halfs

    int bid  = blockIdx.x;              // 2 img * 64 z * 16 y-groups = 2048
    int img  = bid >> 10;
    int rest = bid & 1023;
    int z    = rest >> 4;
    int y0   = (rest & 15) << 2;

    const char* gw = (const char*)wfrag;
    const char* gp = (const char*)(xpad + (long long)img * PV3 * CCH);

    // ---- async stage: weights = 3456 x 16B chunks
    {
        unsigned wbase = lds_byte_off(wlds);
        for (int c = threadIdx.x; c < 3456; c += 256)
            async_load_b128(wbase + c * 16, gw + c * 16);
    }
    // ---- async stage: patch = 18 rows (pz 0..2, py 0..5) x 4224 B = 4752 chunks
    {
        unsigned pbase = lds_byte_off(patch);
        for (int c = threadIdx.x; c < 4752; c += 256) {
            int r   = c / 264;                  // row index 0..17
            int rem = c - r * 264;              // 16B chunk within row
            int pz  = r / 6;
            int py  = r - pz * 6;
            const char* ga = gp
                + ((long long)(((z + pz) * PV) + (y0 + py)) * PV) * (CCH * 2)
                + rem * 16;
            async_load_b128(pbase + c * 16, ga);
        }
    }
    wait_asynccnt0();
    __syncthreads();

    int wave = threadIdx.x >> 5;
    int lane = threadIdx.x & 31;
    int hi   = lane >> 4;
    int n    = lane & 15;

    // two tiles per wave: tile = (yl 0..3, xseg 0..3)
    int t0  = wave * 2;
    int t1  = t0 + 1;
    int yl0 = t0 >> 2, xb0 = (t0 & 3) << 4;
    int yl1 = t1 >> 2, xb1 = (t1 & 3) << 4;

    v8f c00, c01, c10, c11;
    for (int v = 0; v < 8; ++v) {
        float blo = bias[v + 8 * hi];
        float bhi = bias[16 + v + 8 * hi];
        c00[v] = blo; c01[v] = bhi;
        c10[v] = blo; c11[v] = bhi;
    }

#pragma unroll
    for (int T = 0; T < 27; ++T) {
        int dz = T / 9, dy = (T / 3) % 3, dx = T % 3;
        v16bf a0 = *(const v16bf*)&wlds[((T * 2 + 0) * 32 + lane) * 16];
        v16bf a1 = *(const v16bf*)&wlds[((T * 2 + 1) * 32 + lane) * 16];
        v16bf b0 = *(const v16bf*)&patch[((dz * 6 + yl0 + dy) * PV + (xb0 + dx + n)) * CCH + hi * 16];
        v16bf b1 = *(const v16bf*)&patch[((dz * 6 + yl1 + dy) * PV + (xb1 + dx + n)) * CCH + hi * 16];
        c00 = __builtin_amdgcn_wmma_f32_16x16x32_bf16(false, a0, false, b0,
                                                      (short)0, c00, false, false);
        c01 = __builtin_amdgcn_wmma_f32_16x16x32_bf16(false, a1, false, b0,
                                                      (short)0, c01, false, false);
        c10 = __builtin_amdgcn_wmma_f32_16x16x32_bf16(false, a0, false, b1,
                                                      (short)0, c10, false, false);
        c11 = __builtin_amdgcn_wmma_f32_16x16x32_bf16(false, a1, false, b1,
                                                      (short)0, c11, false, false);
    }

    float* o0 = hout + (((long long)img << 18)
                        + ((z * V + (y0 + yl0)) * V + xb0 + n)) * CCH;
    float* o1 = hout + (((long long)img << 18)
                        + ((z * V + (y0 + yl1)) * V + xb1 + n)) * CCH;
#pragma unroll
    for (int v = 0; v < 8; ++v) {
        o0[v + 8 * hi]      = c00[v];
        o0[16 + v + 8 * hi] = c01[v];
        o1[v + 8 * hi]      = c10[v];
        o1[16 + v + 8 * hi] = c11[v];
    }
}

// ---------------------------------------------------------------------------
// 4) GroupNorm statistics over channel-last fp32 [2][64^3][32].
// ---------------------------------------------------------------------------
__global__ void gn_stats_kernel(const float* __restrict__ h,
                                float* __restrict__ stats) {
    __shared__ float ssum[8], ssq[8];
    if (threadIdx.x < 8) { ssum[threadIdx.x] = 0.f; ssq[threadIdx.x] = 0.f; }
    __syncthreads();

    long long idx = (long long)blockIdx.x * blockDim.x + threadIdx.x; // 2*V3
    if (idx < 2LL * V3) {
        int img = (int)(idx >> 18);
        const float4* row = (const float4*)(h + idx * CCH);
        float ls[4] = {0, 0, 0, 0}, lq[4] = {0, 0, 0, 0};
#pragma unroll
        for (int q = 0; q < 8; ++q) {
            float4 f = row[q];
            int g = q >> 1;
            ls[g] += f.x + f.y + f.z + f.w;
            lq[g] += f.x * f.x + f.y * f.y + f.z * f.z + f.w * f.w;
        }
#pragma unroll
        for (int g = 0; g < 4; ++g) {
            atomicAdd(&ssum[img * 4 + g], ls[g]);
            atomicAdd(&ssq[img * 4 + g], lq[g]);
        }
    }
    __syncthreads();
    if (threadIdx.x < 8) {
        atomicAdd(&stats[threadIdx.x * 2],     ssum[threadIdx.x]);
        atomicAdd(&stats[threadIdx.x * 2 + 1], ssq[threadIdx.x]);
    }
}

// ---------------------------------------------------------------------------
// 5) Apply GN + LeakyReLU to h1, write bf16 zero-padded channel-last volume.
// ---------------------------------------------------------------------------
__global__ void gn_lrelu_pad_kernel(const float* __restrict__ h,
                                    const float* __restrict__ stats,
                                    const float* __restrict__ gamma,
                                    const float* __restrict__ beta,
                                    __bf16* __restrict__ xpad) {
    long long idx = (long long)blockIdx.x * blockDim.x + threadIdx.x;
    if (idx >= 2LL * PV3 * CCH) return;
    int ci = (int)(idx & 31);
    long long pv = idx >> 5;
    int img = (int)(pv / PV3);
    int p = (int)(pv % PV3);
    int x = p % PV; int t = p / PV; int y = t % PV; int z = t / PV;
    float val = 0.0f;
    if (x >= 1 && x <= V && y >= 1 && y <= V && z >= 1 && z <= V) {
        long long vox = ((z - 1) * V + (y - 1)) * V + (x - 1);
        float v = h[(((long long)img << 18) + vox) * CCH + ci];
        int g = img * 4 + (ci >> 3);
        float s = stats[g * 2], sq = stats[g * 2 + 1];
        float mean = s * (1.0f / GN_N);
        float var  = sq * (1.0f / GN_N) - mean * mean;
        float xn = (v - mean) * rsqrtf(var + GEPS);
        float o  = xn * gamma[ci] + beta[ci];
        val = (o >= 0.f) ? o : NEG_SLOPE * o;
    }
    xpad[idx] = (__bf16)val;
}

// ---------------------------------------------------------------------------
// 6) Skip-conv (1x1x1) GN statistics: recompute r = Ws*x + bs on the fly.
// ---------------------------------------------------------------------------
__global__ void skip_stats_kernel(const float* __restrict__ moving,
                                  const float* __restrict__ fixedf,
                                  const float* __restrict__ wsw,
                                  const float* __restrict__ bsb,
                                  float* __restrict__ stats) {
    __shared__ float swt[1024];
    __shared__ float ssum[8], ssq[8];
    for (int i = threadIdx.x; i < 1024; i += blockDim.x) swt[i] = wsw[i];
    if (threadIdx.x < 8) { ssum[threadIdx.x] = 0.f; ssq[threadIdx.x] = 0.f; }
    __syncthreads();

    long long idx = (long long)blockIdx.x * blockDim.x + threadIdx.x; // 2*V3
    if (idx < 2LL * V3) {
        int img = (int)(idx >> 18);
        long long vox = idx & (V3 - 1);
        const float* src = img ? fixedf : moving;
        float xv[32];
#pragma unroll
        for (int ci = 0; ci < 32; ++ci) xv[ci] = src[((long long)ci << 18) + vox];
        float ls[4] = {0, 0, 0, 0}, lq[4] = {0, 0, 0, 0};
#pragma unroll
        for (int co = 0; co < 32; ++co) {
            float r = bsb[co];
#pragma unroll
            for (int ci = 0; ci < 32; ++ci) r = fmaf(swt[co * 32 + ci], xv[ci], r);
            int g = co >> 3;
            ls[g] += r; lq[g] += r * r;
        }
#pragma unroll
        for (int g = 0; g < 4; ++g) {
            atomicAdd(&ssum[img * 4 + g], ls[g]);
            atomicAdd(&ssq[img * 4 + g], lq[g]);
        }
    }
    __syncthreads();
    if (threadIdx.x < 8) {
        atomicAdd(&stats[threadIdx.x * 2],     ssum[threadIdx.x]);
        atomicAdd(&stats[threadIdx.x * 2 + 1], ssq[threadIdx.x]);
    }
}

// ---------------------------------------------------------------------------
// 7) Residual finalize: GN(h2) + GN(skip) -> lrelu -> block output (ch-last).
//    stats layout: [0..15]=h1, [16..31]=h2, [32..47]=skip
// ---------------------------------------------------------------------------
__global__ void resid_final_kernel(const float* __restrict__ h2,
                                   const float* __restrict__ stats,
                                   const float* __restrict__ moving,
                                   const float* __restrict__ fixedf,
                                   const float* __restrict__ wsw,
                                   const float* __restrict__ bsb,
                                   const float* __restrict__ g2,
                                   const float* __restrict__ be2,
                                   const float* __restrict__ gsg,
                                   const float* __restrict__ besb,
                                   float* __restrict__ blockout) {
    __shared__ float swt[1024];
    for (int i = threadIdx.x; i < 1024; i += blockDim.x) swt[i] = wsw[i];
    __syncthreads();

    long long idx = (long long)blockIdx.x * blockDim.x + threadIdx.x; // 2*V3
    if (idx >= 2LL * V3) return;
    int img = (int)(idx >> 18);
    long long vox = idx & (V3 - 1);
    const float* src = img ? fixedf : moving;
    float xv[32];
#pragma unroll
    for (int ci = 0; ci < 32; ++ci) xv[ci] = src[((long long)ci << 18) + vox];
    const float* hrow = h2 + idx * CCH;
    float* orow = blockout + idx * CCH;
#pragma unroll
    for (int co = 0; co < 32; ++co) {
        int g = img * 4 + (co >> 3);
        float sh = stats[16 + g * 2], qh = stats[16 + g * 2 + 1];
        float mh = sh * (1.0f / GN_N);
        float vh = qh * (1.0f / GN_N) - mh * mh;
        float hn = (hrow[co] - mh) * rsqrtf(vh + GEPS) * g2[co] + be2[co];
        float r = bsb[co];
#pragma unroll
        for (int ci = 0; ci < 32; ++ci) r = fmaf(swt[co * 32 + ci], xv[ci], r);
        float sr = stats[32 + g * 2], qr = stats[32 + g * 2 + 1];
        float mr = sr * (1.0f / GN_N);
        float vr = qr * (1.0f / GN_N) - mr * mr;
        float rn = (r - mr) * rsqrtf(vr + GEPS) * gsg[co] + besb[co];
        float o = hn + rn;
        orow[co] = (o >= 0.f) ? o : NEG_SLOPE * o;
    }
}

// ---------------------------------------------------------------------------
// 8) Projection 32->16 for fixed image (Q, unpadded 64^3, channel-last)
// ---------------------------------------------------------------------------
__global__ void proj_q_kernel(const float* __restrict__ blockout,
                              const float* __restrict__ wp,
                              const float* __restrict__ bp,
                              float* __restrict__ Q) {
    __shared__ float swp[512];
    for (int i = threadIdx.x; i < 512; i += blockDim.x) swp[i] = wp[i];
    __syncthreads();
    long long idx = (long long)blockIdx.x * blockDim.x + threadIdx.x; // V3
    if (idx >= V3) return;
    const float* row = blockout + ((1LL << 18) + idx) * CCH;   // img 1 = fixed
    float xv[32];
#pragma unroll
    for (int ci = 0; ci < 32; ++ci) xv[ci] = row[ci];
#pragma unroll
    for (int c = 0; c < 16; ++c) {
        float s = bp[c];
#pragma unroll
        for (int ci = 0; ci < 32; ++ci) s = fmaf(swp[c * 32 + ci], xv[ci], s);
        Q[idx * CEMB + c] = s;
    }
}

// ---------------------------------------------------------------------------
// 9) Projection 32->16 for moving image with edge-clamp padding (66^3 grid)
// ---------------------------------------------------------------------------
__global__ void proj_emb_kernel(const float* __restrict__ blockout,
                                const float* __restrict__ wp,
                                const float* __restrict__ bp,
                                float* __restrict__ emb) {
    __shared__ float swp[512];
    for (int i = threadIdx.x; i < 512; i += blockDim.x) swp[i] = wp[i];
    __syncthreads();
    long long idx = (long long)blockIdx.x * blockDim.x + threadIdx.x; // PV3
    if (idx >= PV3) return;
    int p = (int)idx;
    int x = p % PV; int t = p / PV; int y = t % PV; int z = t / PV;
    int cx = min(max(x - 1, 0), V - 1);
    int cy = min(max(y - 1, 0), V - 1);
    int cz = min(max(z - 1, 0), V - 1);
    const float* row = blockout + (long long)((cz * V + cy) * V + cx) * CCH; // img 0
    float xv[32];
#pragma unroll
    for (int ci = 0; ci < 32; ++ci) xv[ci] = row[ci];
#pragma unroll
    for (int c = 0; c < 16; ++c) {
        float s = bp[c];
#pragma unroll
        for (int ci = 0; ci < 32; ++ci) s = fmaf(swp[c * 32 + ci], xv[ci], s);
        emb[idx * CEMB + c] = s;
    }
}

// ---------------------------------------------------------------------------
// 10) 27-neighbor attention -> radial displacement field [3][64^3]
// ---------------------------------------------------------------------------
__global__ void attn_kernel(const float* __restrict__ Q,
                            const float* __restrict__ emb,
                            float* __restrict__ out) {
    long long idx = (long long)blockIdx.x * blockDim.x + threadIdx.x; // V3
    if (idx >= V3) return;
    int p = (int)idx;
    int x = p % V; int t = p / V; int y = t % V; int z = t / V;

    float q[16];
#pragma unroll
    for (int c = 0; c < 16; ++c) q[c] = Q[idx * CEMB + c];

    float sc[27];
    float mx = -3.402823466e+38f;
#pragma unroll
    for (int pp = 0; pp < 27; ++pp) {
        int dz = pp / 9, dy = (pp / 3) % 3, dx = pp % 3;
        const float* k = emb + (long long)(((z + dz) * PV + (y + dy)) * PV + (x + dx)) * CEMB;
        float s = 0.f;
#pragma unroll
        for (int c = 0; c < 16; ++c) s = fmaf(q[c], k[c], s);
        s *= 0.25f;                        // 1/sqrt(16)
        sc[pp] = s;
        mx = fmaxf(mx, s);
    }
    float se = 0.f, d0 = 0.f, d1 = 0.f, d2 = 0.f;
#pragma unroll
    for (int pp = 0; pp < 27; ++pp) {
        float e = __expf(sc[pp] - mx);
        se += e;
        d0 += e * (float)((pp / 9) - 1);
        d1 += e * (float)(((pp / 3) % 3) - 1);
        d2 += e * (float)((pp % 3) - 1);
    }
    float inv = 1.0f / se;
    out[idx]            = d0 * inv;
    out[V3 + idx]       = d1 * inv;
    out[2LL * V3 + idx] = d2 * inv;
}

// ---------------------------------------------------------------------------
// Host-side orchestration
// ---------------------------------------------------------------------------
extern "C" void kernel_launch(void* const* d_in, const int* in_sizes, int n_in,
                              void* d_out, int out_size, void* d_ws, size_t ws_size,
                              hipStream_t stream) {
    const float* feat_moving = (const float*)d_in[0];
    const float* feat_fixed  = (const float*)d_in[1];
    const float* w1  = (const float*)d_in[2];
    const float* b1  = (const float*)d_in[3];
    const float* g1  = (const float*)d_in[4];
    const float* be1 = (const float*)d_in[5];
    const float* w2  = (const float*)d_in[6];
    const float* b2  = (const float*)d_in[7];
    const float* g2  = (const float*)d_in[8];
    const float* be2 = (const float*)d_in[9];
    const float* wsw = (const float*)d_in[10];
    const float* bsb = (const float*)d_in[11];
    const float* gsg = (const float*)d_in[12];
    const float* besb= (const float*)d_in[13];
    const float* wp  = (const float*)d_in[14];
    const float* bp  = (const float*)d_in[15];
    float* out = (float*)d_out;

    char* wsp = (char*)d_ws;
    size_t off = 0;
    auto alloc = [&](size_t bytes) -> size_t {
        size_t cur = off;
        off = (off + bytes + 255) & ~(size_t)255;
        return cur;
    };
    float*  stats    = (float*)(wsp + alloc(192));                       // 48 f32
    __bf16* wfrag    = (__bf16*)(wsp + alloc(27648 * 2));                // 55 KB
    __bf16* xpad     = (__bf16*)(wsp + alloc(2ULL * PV3 * CCH * 2));     // 36.8 MB
    float*  hbuf     = (float*)(wsp + alloc(2ULL * V3 * CCH * 4));       // 67 MB
    float*  blockout = (float*)(wsp + alloc(2ULL * V3 * CCH * 4));       // 67 MB
    float*  Qbuf     = (float*)(wsp + alloc((size_t)V3 * CEMB * 4));     // 16.8 MB
    float*  embbuf   = (float*)(wsp + alloc((size_t)PV3 * CEMB * 4));    // 18.4 MB
    (void)ws_size; (void)in_sizes; (void)n_in; (void)out_size;

    const int padBlocks  = (int)((2LL * PV3 * CCH + 255) / 256);   // 71874
    const int voxBlocks  = (int)((2LL * V3 + 255) / 256);          // 2048
    const int convBlocks = 2 * 64 * 16;                            // 2048
    const int qBlocks    = (V3 + 255) / 256;                       // 1024
    const int embBlocks  = (PV3 + 255) / 256;                      // 1124

    hipMemsetAsync(stats, 0, 192, stream);

    // cast + zero-pad both images to bf16 channel-last
    cast_pad_kernel<<<padBlocks, 256, 0, stream>>>(feat_moving, feat_fixed, xpad);

    // conv1 (WMMA, LDS-staged) + GN stats + GN/lrelu -> bf16 padded
    prep_weights_kernel<<<108, 256, 0, stream>>>(w1, wfrag);
    conv3_wmma_kernel<<<convBlocks, 256, CONV_SMEM_BYTES, stream>>>(xpad, wfrag, b1, hbuf);
    gn_stats_kernel<<<voxBlocks, 256, 0, stream>>>(hbuf, stats);           // -> [0..15]
    gn_lrelu_pad_kernel<<<padBlocks, 256, 0, stream>>>(hbuf, stats, g1, be1, xpad);

    // conv2 (WMMA, LDS-staged) + GN stats
    prep_weights_kernel<<<108, 256, 0, stream>>>(w2, wfrag);
    conv3_wmma_kernel<<<convBlocks, 256, CONV_SMEM_BYTES, stream>>>(xpad, wfrag, b2, hbuf);
    gn_stats_kernel<<<voxBlocks, 256, 0, stream>>>(hbuf, stats + 16);      // -> [16..31]

    // skip conv stats, then residual finalize
    skip_stats_kernel<<<voxBlocks, 256, 0, stream>>>(feat_moving, feat_fixed,
                                                     wsw, bsb, stats + 32);// -> [32..47]
    resid_final_kernel<<<voxBlocks, 256, 0, stream>>>(hbuf, stats,
                                                      feat_moving, feat_fixed,
                                                      wsw, bsb, g2, be2, gsg, besb,
                                                      blockout);

    // projections + local attention
    proj_q_kernel<<<qBlocks, 256, 0, stream>>>(blockout, wp, bp, Qbuf);
    proj_emb_kernel<<<embBlocks, 256, 0, stream>>>(blockout, wp, bp, embbuf);
    attn_kernel<<<qBlocks, 256, 0, stream>>>(Qbuf, embbuf, out);
}
